// Qwen3Experts_10565619548609
// MI455X (gfx1250) — matrix operand order repaired and checked by reading, not compiled, not run on hardware
//
#include <hip/hip_runtime.h>
#include <cstdint>
#include <cstddef>

// ---------------------------------------------------------------------------
// Qwen3-style MoE experts for MI455X (gfx1250, wave32, WMMA).
//   Weights stay fp32 in HBM (1.2 GB ~= 52us @ 23.3 TB/s) and are converted
//   to bf16 in-flight while staging to LDS; all GEMMs run on
//   v_wmma_f32_16x16x32_bf16. The down-proj A operand is already bf16 in the
//   workspace and is staged with gfx1250 async global->LDS copies.
//   Work is dispatched through a compact (expert, rowTile) list built on
//   device (<=320 tiles) instead of a mostly-empty 3D grid.
// ---------------------------------------------------------------------------

#define NEXPERTS 64
#define TOPK     8
#define HIDDEN   2048
#define INTER    768
#define TOKENS   4096
#define ROWS     (TOKENS * TOPK)   // 32768 (token, expert) pairs

#define BM  128                    // rows per block tile
#define BN  64                     // output cols per block tile
#define BK  32                     // K slice per stage (one WMMA K)
#define LDK 48                     // padded LDS K-stride: 96B rows, 16B aligned
#define MAXTILES (ROWS / BM + NEXPERTS)   // 320 worst-case row tiles

typedef __attribute__((ext_vector_type(16))) __bf16 bf16x16;
typedef __attribute__((ext_vector_type(8)))  __bf16 bf16x8;
typedef __attribute__((ext_vector_type(2)))  __bf16 bf16x2;
typedef __attribute__((ext_vector_type(8)))  float  f32x8;
typedef __attribute__((ext_vector_type(4)))  int    i32x4;

#if defined(__has_builtin)
#if __has_builtin(__builtin_amdgcn_global_load_async_to_lds_b128)
#define HAVE_ASYNC_COPY 1
#endif
#if __has_builtin(__builtin_amdgcn_s_wait_asynccnt)
#define HAVE_WAIT_ASYNC 1
#endif
#endif

__device__ __forceinline__ void wait_async0() {
#if defined(HAVE_WAIT_ASYNC)
  __builtin_amdgcn_s_wait_asynccnt(0);
#elif defined(HAVE_ASYNC_COPY)
  asm volatile("s_wait_asynccnt 0" ::: "memory");
#endif
}

// A fragment (16x32 bf16, ISA 7.12.2): lane<16 -> row=lane, K {0..7,16..23};
// lane>=16 -> row=lane-16, K {8..15,24..31}. Two aligned ds_load_b128.
__device__ __forceinline__ bf16x16 frag_a(const __bf16* tile, int lane) {
  const int half = lane >> 4, lr = lane & 15;
  const __bf16* p = tile + lr * LDK + half * 8;
  bf16x8 lo = *(const bf16x8*)(p);
  bf16x8 hi = *(const bf16x8*)(p + 16);
  bf16x16 r;
#pragma unroll
  for (int i = 0; i < 8; ++i) { r[i] = lo[i]; r[i + 8] = hi[i]; }
  return r;
}

// B fragment (32x16 bf16): lane holds column N=lane&15; K {0..7,16..23}
// (lane<16) or {8..15,24..31} (lane>=16). LDS holds B column-major [n][k].
__device__ __forceinline__ bf16x16 frag_b(const __bf16* tile, int lane) {
  const int half = lane >> 4, lc = lane & 15;
  const __bf16* p = tile + lc * LDK + half * 8;
  bf16x8 lo = *(const bf16x8*)(p);
  bf16x8 hi = *(const bf16x8*)(p + 16);
  bf16x16 r;
#pragma unroll
  for (int i = 0; i < 8; ++i) { r[i] = lo[i]; r[i + 8] = hi[i]; }
  return r;
}

__device__ __forceinline__ f32x8 wmma_bf16(bf16x16 a, bf16x16 b, f32x8 c) {
  return __builtin_amdgcn_wmma_f32_16x16x32_bf16(false, a, false, b,
                                                 (short)0, c, false, false);
}

// ---------------------------------------------------------------------------
// Routing: per-token top-8 + softmax; count rows per expert.
// ---------------------------------------------------------------------------
__global__ void moe_router(const float* __restrict__ logits,
                           int* __restrict__ selExpert,
                           float* __restrict__ selWeight,
                           int* __restrict__ counts) {
  int t = blockIdx.x * blockDim.x + threadIdx.x;
  if (t >= TOKENS) return;
  float v[NEXPERTS];
#pragma unroll
  for (int i = 0; i < NEXPERTS; ++i) v[i] = logits[t * NEXPERTS + i];
  float wk[TOPK];
  int ek[TOPK];
  for (int k = 0; k < TOPK; ++k) {
    float best = -3.4e38f; int bi = 0;
    for (int i = 0; i < NEXPERTS; ++i)
      if (v[i] > best) { best = v[i]; bi = i; }
    wk[k] = best; ek[k] = bi; v[bi] = -3.4e38f;
  }
  float m = wk[0], s = 0.f, p[TOPK];
#pragma unroll
  for (int k = 0; k < TOPK; ++k) { p[k] = __expf(wk[k] - m); s += p[k]; }
  float inv = 1.f / s;
#pragma unroll
  for (int k = 0; k < TOPK; ++k) {
    selExpert[t * TOPK + k] = ek[k];
    selWeight[t * TOPK + k] = p[k] * inv;
    atomicAdd(&counts[ek[k]], 1);
  }
}

// Exclusive scan over expert counts + compact (expert, rowTile) work list.
__global__ void moe_scan(const int* __restrict__ counts,
                         int* __restrict__ offsets, int* __restrict__ cursor,
                         int* __restrict__ tileExpert,
                         int* __restrict__ tileRow, int* __restrict__ nTiles) {
  if (threadIdx.x == 0) {
    int o = 0, tt = 0;
    for (int e = 0; e < NEXPERTS; ++e) {
      offsets[e] = o;
      int c = counts[e];
      o += c;
      for (int r = 0; r < c; r += BM) { tileExpert[tt] = e; tileRow[tt] = r; ++tt; }
    }
    offsets[NEXPERTS] = o;
    *nTiles = tt;
  }
  if (threadIdx.x < NEXPERTS) cursor[threadIdx.x] = 0;
}

__global__ void moe_scatter(const int* __restrict__ selExpert,
                            const float* __restrict__ selWeight,
                            const int* __restrict__ offsets,
                            int* __restrict__ cursor,
                            int* __restrict__ rowToken,
                            float* __restrict__ rowWeight) {
  int i = blockIdx.x * blockDim.x + threadIdx.x;
  if (i >= ROWS) return;
  int e = selExpert[i];
  int dst = offsets[e] + atomicAdd(&cursor[e], 1);
  rowToken[dst]  = i >> 3;           // token id
  rowWeight[dst] = selWeight[i];
}

// ---------------------------------------------------------------------------
// Grouped GEMM 1: act = silu(hs @ gate) * (hs @ up), bf16 out to workspace.
// Block: 256 threads = 8 waves; wave (wm,wn) owns 32x32 of the 128x64 tile.
// ---------------------------------------------------------------------------
__global__ __launch_bounds__(256) void moe_gateup(
    const float* __restrict__ hs, const float* __restrict__ gate,
    const float* __restrict__ up, const int* __restrict__ counts,
    const int* __restrict__ offsets, const int* __restrict__ rowToken,
    const int* __restrict__ tileExpert, const int* __restrict__ tileRow,
    const int* __restrict__ nTiles, __bf16* __restrict__ act) {
  const int ti = blockIdx.y;
  if (ti >= *nTiles) return;
  const int e = tileExpert[ti];
  const int rowBase = tileRow[ti];
  const int cnt = counts[e];
  const int nBase = blockIdx.x * BN;
  const int base = offsets[e];

  __shared__ __bf16 sA[2][BM][LDK];
  __shared__ __bf16 sG[2][BN][LDK];
  __shared__ __bf16 sU[2][BN][LDK];

  const int tid = threadIdx.x;
  const int lane = tid & 31, wave = tid >> 5;
  const int wm = wave & 3, wn = wave >> 2;

  // A staging: 128 rows x 4 chunks of 8 floats = 512, 2 per thread.
  int aRow[2], aO8[2], aTok[2];
#pragma unroll
  for (int c = 0; c < 2; ++c) {
    int lin = tid + 256 * c;
    aRow[c] = lin >> 2;
    aO8[c]  = (lin & 3) * 8;
    int gr = rowBase + aRow[c];
    aTok[c] = (gr < cnt) ? rowToken[base + gr] : -1;
  }
  // B staging: 16 k-pairs x 16 n-quads = 256 double-chunks, 1 per thread.
  const int kp2 = (tid >> 4) * 2;
  const int n0  = (tid & 15) * 4;

  float4 ra[2][2], rg[2], ru[2];
  auto loadRegs = [&](int k0) {
#pragma unroll
    for (int c = 0; c < 2; ++c) {
      if (aTok[c] >= 0) {
        const float* src = hs + (size_t)aTok[c] * HIDDEN + k0 + aO8[c];
        ra[c][0] = *(const float4*)(src);
        ra[c][1] = *(const float4*)(src + 4);
      } else {
        ra[c][0] = make_float4(0.f, 0.f, 0.f, 0.f);
        ra[c][1] = make_float4(0.f, 0.f, 0.f, 0.f);
      }
    }
    size_t off = ((size_t)e * HIDDEN + (size_t)(k0 + kp2)) * INTER + nBase + n0;
    rg[0] = *(const float4*)(gate + off);
    rg[1] = *(const float4*)(gate + off + INTER);
    ru[0] = *(const float4*)(up + off);
    ru[1] = *(const float4*)(up + off + INTER);
  };
  auto storeLds = [&](int s) {
#pragma unroll
    for (int c = 0; c < 2; ++c) {
      bf16x8 v;
      v[0] = (__bf16)ra[c][0].x; v[1] = (__bf16)ra[c][0].y;
      v[2] = (__bf16)ra[c][0].z; v[3] = (__bf16)ra[c][0].w;
      v[4] = (__bf16)ra[c][1].x; v[5] = (__bf16)ra[c][1].y;
      v[6] = (__bf16)ra[c][1].z; v[7] = (__bf16)ra[c][1].w;
      *(bf16x8*)&sA[s][aRow[c]][aO8[c]] = v;
    }
    bf16x2 p;  // transpose to [n][k]; k-pairs pack into b32 stores
    p[0] = (__bf16)rg[0].x; p[1] = (__bf16)rg[1].x; *(bf16x2*)&sG[s][n0 + 0][kp2] = p;
    p[0] = (__bf16)rg[0].y; p[1] = (__bf16)rg[1].y; *(bf16x2*)&sG[s][n0 + 1][kp2] = p;
    p[0] = (__bf16)rg[0].z; p[1] = (__bf16)rg[1].z; *(bf16x2*)&sG[s][n0 + 2][kp2] = p;
    p[0] = (__bf16)rg[0].w; p[1] = (__bf16)rg[1].w; *(bf16x2*)&sG[s][n0 + 3][kp2] = p;
    p[0] = (__bf16)ru[0].x; p[1] = (__bf16)ru[1].x; *(bf16x2*)&sU[s][n0 + 0][kp2] = p;
    p[0] = (__bf16)ru[0].y; p[1] = (__bf16)ru[1].y; *(bf16x2*)&sU[s][n0 + 1][kp2] = p;
    p[0] = (__bf16)ru[0].z; p[1] = (__bf16)ru[1].z; *(bf16x2*)&sU[s][n0 + 2][kp2] = p;
    p[0] = (__bf16)ru[0].w; p[1] = (__bf16)ru[1].w; *(bf16x2*)&sU[s][n0 + 3][kp2] = p;
  };

  f32x8 zero = {};
  f32x8 accG[2][2], accU[2][2];
#pragma unroll
  for (int mi = 0; mi < 2; ++mi)
#pragma unroll
    for (int ni = 0; ni < 2; ++ni) { accG[mi][ni] = zero; accU[mi][ni] = zero; }

  loadRegs(0);
  storeLds(0);
  __syncthreads();

  const int KS = HIDDEN / BK;  // 64
  for (int ks = 0; ks < KS; ++ks) {
    const int cur = ks & 1, nxt = cur ^ 1;
    const bool more = (ks + 1 < KS);
    if (more) loadRegs((ks + 1) * BK);

    bf16x16 af[2], bg[2], bu[2];
#pragma unroll
    for (int mi = 0; mi < 2; ++mi)
      af[mi] = frag_a(&sA[cur][wm * 32 + mi * 16][0], lane);
#pragma unroll
    for (int ni = 0; ni < 2; ++ni) {
      bg[ni] = frag_b(&sG[cur][wn * 32 + ni * 16][0], lane);
      bu[ni] = frag_b(&sU[cur][wn * 32 + ni * 16][0], lane);
    }
#pragma unroll
    for (int mi = 0; mi < 2; ++mi)
#pragma unroll
      for (int ni = 0; ni < 2; ++ni) {
        accG[mi][ni] = wmma_bf16(af[mi], bg[ni], accG[mi][ni]);
        accU[mi][ni] = wmma_bf16(af[mi], bu[ni], accU[mi][ni]);
      }
    if (more) { storeLds(nxt); __syncthreads(); }
  }

  // SwiGLU epilogue -> bf16 intermediate. C layout: lane<16 N=lane, M=vgpr;
  // lane>=16 N=lane-16, M=vgpr+8.
  const int half = lane >> 4, ln = lane & 15;
#pragma unroll
  for (int mi = 0; mi < 2; ++mi)
#pragma unroll
    for (int ni = 0; ni < 2; ++ni) {
      int col = nBase + wn * 32 + ni * 16 + ln;
#pragma unroll
      for (int i = 0; i < 8; ++i) {
        int gr = rowBase + wm * 32 + mi * 16 + half * 8 + i;
        if (gr < cnt) {
          float g = accG[mi][ni][i], u = accU[mi][ni][i];
          float a = (g / (1.f + __expf(-g))) * u;
          act[(size_t)(base + gr) * INTER + col] = (__bf16)a;
        }
      }
    }
}

// ---------------------------------------------------------------------------
// Grouped GEMM 2: out[token] += w * (act @ down). A staged with async
// global->LDS copies (bf16 already), B converted fp32->bf16 via VGPRs.
// ---------------------------------------------------------------------------
__global__ __launch_bounds__(256) void moe_down(
    const __bf16* __restrict__ act, const float* __restrict__ down,
    const int* __restrict__ counts, const int* __restrict__ offsets,
    const int* __restrict__ rowToken, const float* __restrict__ rowWeight,
    const int* __restrict__ tileExpert, const int* __restrict__ tileRow,
    const int* __restrict__ nTiles, float* __restrict__ out) {
  const int ti = blockIdx.y;
  if (ti >= *nTiles) return;
  const int e = tileExpert[ti];
  const int rowBase = tileRow[ti];
  const int cnt = counts[e];
  const int nBase = blockIdx.x * BN;
  const int base = offsets[e];

  __shared__ __bf16 sA[2][BM][LDK];
  __shared__ __bf16 sB[2][BN][LDK];

  const int tid = threadIdx.x;
  const int lane = tid & 31, wave = tid >> 5;
  const int wm = wave & 3, wn = wave >> 2;

  // A: 128 rows x 4 x 16B chunks = 512, 2 per thread.
  int aRow[2], aQ[2];
  const __bf16* aSrc[2];
#pragma unroll
  for (int c = 0; c < 2; ++c) {
    int lin = tid + 256 * c;
    aRow[c] = lin >> 2;
    aQ[c]   = lin & 3;
    int gr = rowBase + aRow[c];
    if (gr >= cnt) gr = cnt - 1;  // clamp; those output rows are discarded
    aSrc[c] = act + (size_t)(base + gr) * INTER + aQ[c] * 8;
  }
  const int kp2 = (tid >> 4) * 2;
  const int n0  = (tid & 15) * 4;

  auto asyncA = [&](int s, int k0) {
#pragma unroll
    for (int c = 0; c < 2; ++c) {
#if defined(HAVE_ASYNC_COPY)
      // src: v4i in AS(1), dst: v4i in AS(3), imm offset, imm cpol.
      __builtin_amdgcn_global_load_async_to_lds_b128(
          (__attribute__((address_space(1))) i32x4*)(aSrc[c] + k0),
          (__attribute__((address_space(3))) i32x4*)&sA[s][aRow[c]][aQ[c] * 8],
          0, 0);
#else
      *(bf16x8*)&sA[s][aRow[c]][aQ[c] * 8] = *(const bf16x8*)(aSrc[c] + k0);
#endif
    }
  };
  float4 rb[2];
  auto loadB = [&](int k0) {
    size_t off = ((size_t)e * INTER + (size_t)(k0 + kp2)) * HIDDEN + nBase + n0;
    rb[0] = *(const float4*)(down + off);
    rb[1] = *(const float4*)(down + off + HIDDEN);
  };
  auto storeB = [&](int s) {
    bf16x2 p;
    p[0] = (__bf16)rb[0].x; p[1] = (__bf16)rb[1].x; *(bf16x2*)&sB[s][n0 + 0][kp2] = p;
    p[0] = (__bf16)rb[0].y; p[1] = (__bf16)rb[1].y; *(bf16x2*)&sB[s][n0 + 1][kp2] = p;
    p[0] = (__bf16)rb[0].z; p[1] = (__bf16)rb[1].z; *(bf16x2*)&sB[s][n0 + 2][kp2] = p;
    p[0] = (__bf16)rb[0].w; p[1] = (__bf16)rb[1].w; *(bf16x2*)&sB[s][n0 + 3][kp2] = p;
  };

  f32x8 zero = {};
  f32x8 accD[2][2];
#pragma unroll
  for (int mi = 0; mi < 2; ++mi)
#pragma unroll
    for (int ni = 0; ni < 2; ++ni) accD[mi][ni] = zero;

  asyncA(0, 0);
  loadB(0);
  storeB(0);
  wait_async0();
  __syncthreads();

  const int KS = INTER / BK;  // 24
  for (int ks = 0; ks < KS; ++ks) {
    const int cur = ks & 1, nxt = cur ^ 1;
    const bool more = (ks + 1 < KS);
    if (more) { loadB((ks + 1) * BK); asyncA(nxt, (ks + 1) * BK); }

    bf16x16 af[2], bf[2];
#pragma unroll
    for (int mi = 0; mi < 2; ++mi)
      af[mi] = frag_a(&sA[cur][wm * 32 + mi * 16][0], lane);
#pragma unroll
    for (int ni = 0; ni < 2; ++ni)
      bf[ni] = frag_b(&sB[cur][wn * 32 + ni * 16][0], lane);
#pragma unroll
    for (int mi = 0; mi < 2; ++mi)
#pragma unroll
      for (int ni = 0; ni < 2; ++ni)
        accD[mi][ni] = wmma_bf16(af[mi], bf[ni], accD[mi][ni]);

    if (more) { storeB(nxt); wait_async0(); __syncthreads(); }
  }

  // Routed combine: out[token, col] += weight * d  (global_atomic_add_f32).
  const int half = lane >> 4, ln = lane & 15;
#pragma unroll
  for (int mi = 0; mi < 2; ++mi)
#pragma unroll
    for (int ni = 0; ni < 2; ++ni) {
      int col = nBase + wn * 32 + ni * 16 + ln;
#pragma unroll
      for (int i = 0; i < 8; ++i) {
        int gr = rowBase + wm * 32 + mi * 16 + half * 8 + i;
        if (gr < cnt) {
          int r = base + gr;
          unsafeAtomicAdd(out + (size_t)rowToken[r] * HIDDEN + col,
                          rowWeight[r] * accD[mi][ni][i]);
        }
      }
    }
}

// ---------------------------------------------------------------------------
// Host launch. Workspace (~51 MB): bf16 act [ROWS x INTER] + routing metadata.
// ---------------------------------------------------------------------------
extern "C" void kernel_launch(void* const* d_in, const int* in_sizes, int n_in,
                              void* d_out, int out_size, void* d_ws,
                              size_t ws_size, hipStream_t stream) {
  (void)in_sizes; (void)n_in; (void)ws_size;
  const float* hs     = (const float*)d_in[0];
  const float* logits = (const float*)d_in[1];
  const float* gate   = (const float*)d_in[2];
  const float* up     = (const float*)d_in[3];
  const float* down   = (const float*)d_in[4];
  float* out = (float*)d_out;

  char* p = (char*)d_ws;
  __bf16* act       = (__bf16*)p;  p += (size_t)ROWS * INTER * sizeof(__bf16);
  int*   selExpert  = (int*)p;     p += (size_t)ROWS * sizeof(int);
  float* selWeight  = (float*)p;   p += (size_t)ROWS * sizeof(float);
  int*   rowToken   = (int*)p;     p += (size_t)ROWS * sizeof(int);
  float* rowWeight  = (float*)p;   p += (size_t)ROWS * sizeof(float);
  int*   counts     = (int*)p;     p += NEXPERTS * sizeof(int);
  int*   offsets    = (int*)p;     p += (NEXPERTS + 1) * sizeof(int);
  int*   cursor     = (int*)p;     p += NEXPERTS * sizeof(int);
  int*   tileExpert = (int*)p;     p += MAXTILES * sizeof(int);
  int*   tileRow    = (int*)p;     p += MAXTILES * sizeof(int);
  int*   nTiles     = (int*)p;     p += sizeof(int);

  (void)hipMemsetAsync(out, 0, (size_t)out_size * sizeof(float), stream);
  (void)hipMemsetAsync(counts, 0, (3 * NEXPERTS + 1) * sizeof(int), stream);

  moe_router<<<TOKENS / 128, 128, 0, stream>>>(logits, selExpert, selWeight,
                                               counts);
  moe_scan<<<1, 64, 0, stream>>>(counts, offsets, cursor, tileExpert, tileRow,
                                 nTiles);
  moe_scatter<<<ROWS / 256, 256, 0, stream>>>(selExpert, selWeight, offsets,
                                              cursor, rowToken, rowWeight);
  moe_gateup<<<dim3(INTER / BN, MAXTILES), 256, 0, stream>>>(
      hs, gate, up, counts, offsets, rowToken, tileExpert, tileRow, nTiles,
      act);
  moe_down<<<dim3(HIDDEN / BN, MAXTILES), 256, 0, stream>>>(
      act, down, counts, offsets, rowToken, rowWeight, tileExpert, tileRow,
      nTiles, out);
}
